// CrossAttention_43714177139148
// MI455X (gfx1250) — matrix operand order repaired and checked by reading, not compiled, not run on hardware
//
#include <hip/hip_runtime.h>
#include <hip/hip_bf16.h>

// Problem constants (from reference)
constexpr int B  = 2;
constexpr int NQ = 2048;
constexpr int NK = 2048;
constexpr int D  = 1024;
constexpr int VD = 512;
constexpr int H  = 16;
constexpr int HD = D / H;   // 64

typedef __attribute__((ext_vector_type(16))) __bf16        v16bf;
typedef __attribute__((ext_vector_type(8)))  __bf16        v8bf;
typedef __attribute__((ext_vector_type(8)))  float         v8f;
typedef __attribute__((ext_vector_type(4)))  unsigned int  v4u;
typedef __attribute__((ext_vector_type(4)))  int           v4i;
typedef __attribute__((ext_vector_type(8)))  int           v8i;

#define HAVE_TDM __has_builtin(__builtin_amdgcn_tensor_load_to_lds)

// ---------------- helpers ----------------
__device__ __forceinline__ unsigned short f2bf(float f) {
  union { float f; unsigned u; } v; v.f = f;
  unsigned r = v.u + 0x7FFFu + ((v.u >> 16) & 1u);   // round-to-nearest-even
  return (unsigned short)(r >> 16);
}

__device__ __forceinline__ float warp16_max(float v) {
  #pragma unroll
  for (int m = 1; m < 16; m <<= 1) v = fmaxf(v, __shfl_xor(v, m, 32));
  return v;
}
__device__ __forceinline__ float warp16_sum(float v) {
  #pragma unroll
  for (int m = 1; m < 16; m <<= 1) v += __shfl_xor(v, m, 32);
  return v;
}

// A fragment (16-bit, 16x32): lane holds row (lane&15); lanes<16 get K
// [0..15]; lanes>=16 get [16..31]. loA = (lane<16)?0:16.
__device__ __forceinline__ v16bf load_a_bf16(const unsigned short* rowp, int k, int loA) {
  v16bf a;
  ((v8bf*)&a)[0] = *(const v8bf*)(rowp + k + loA);
  ((v8bf*)&a)[1] = *(const v8bf*)(rowp + k + loA + 8);
  return a;
}

#if HAVE_TDM
// ---- Tensor Data Mover: 2D bf16 tile (tile_w x tile_h elems) global -> LDS ----
// D# per CDNA5 ISA ch.8: group0 = {flags, lds_addr, global_addr, type=2},
// group1 = {data_size=2B, tensor dims, tile dims, dim0 stride}. Groups 2/3 zero (2D).
__device__ __forceinline__ void tdm_load_2d_bf16(unsigned lds_off, const void* gptr,
                                                 unsigned tile_w, unsigned tile_h,
                                                 unsigned row_stride_elems) {
  unsigned long long ga = (unsigned long long)(uintptr_t)gptr;
  unsigned ga_lo = (unsigned)__builtin_amdgcn_readfirstlane((int)(unsigned)ga);
  unsigned ga_hi = (unsigned)__builtin_amdgcn_readfirstlane((int)(unsigned)(ga >> 32));
  unsigned ldso  = (unsigned)__builtin_amdgcn_readfirstlane((int)lds_off);
  unsigned strd  = (unsigned)__builtin_amdgcn_readfirstlane((int)row_stride_elems);

  v4u g0;
  g0[0] = 1u;                                   // count=1, is_restore=0, no gather
  g0[1] = ldso;                                 // lds_addr (bytes)
  g0[2] = ga_lo;                                // global_addr[31:0]
  g0[3] = (ga_hi & 0x01FFFFFFu) | (2u << 30);   // global_addr[56:32] | type=2

  v8i g1;
  g1[0] = (int)(1u << 16);                          // wg_mask=0, data_size=1 (2 bytes)
  g1[1] = (int)((tile_w & 0xFFFFu) << 16);          // tensor_dim0[15:0]
  g1[2] = (int)((tile_w >> 16) & 0xFFFFu) | (int)((tile_h & 0xFFFFu) << 16); // dim0 hi | dim1 lo
  g1[3] = (int)((tile_h >> 16) & 0xFFFFu) | (int)((tile_w & 0xFFFFu) << 16); // dim1 hi | tile_dim0
  g1[4] = (int)(tile_h & 0xFFFFu);                  // tile_dim1; tile_dim2=0
  g1[5] = (int)strd;                                // tensor_dim0_stride[31:0]
  g1[6] = 0;                                        // stride0 hi | stride1 lo
  g1[7] = 0;                                        // stride1 hi

  v4i gz = {0, 0, 0, 0};
#if __clang_major__ >= 23
  v8i gz8 = {0, 0, 0, 0, 0, 0, 0, 0};
  __builtin_amdgcn_tensor_load_to_lds(g0, g1, gz, gz, gz8, 0);
#else
  __builtin_amdgcn_tensor_load_to_lds(g0, g1, gz, gz, 0);
#endif
}
#endif // HAVE_TDM

// ---------------- kernel 1: RMSNorm rows -> bf16 ----------------
__global__ void rmsnorm_bf16_kernel(const float* __restrict__ x,
                                    const float* __restrict__ w,
                                    unsigned short* __restrict__ out) {
  int row = blockIdx.x;
  const float* xr = x + (size_t)row * D;
  __shared__ float red[256];
  float s = 0.f;
  for (int i = threadIdx.x; i < D; i += 256) { float v = xr[i]; s += v * v; }
  red[threadIdx.x] = s; __syncthreads();
  for (int off = 128; off > 0; off >>= 1) {
    if (threadIdx.x < off) red[threadIdx.x] += red[threadIdx.x + off];
    __syncthreads();
  }
  float inv = rsqrtf(red[0] * (1.0f / D) + 1.1920929e-07f);
  for (int i = threadIdx.x; i < D; i += 256)
    out[(size_t)row * D + i] = f2bf(xr[i] * inv * w[i]);
}

// ---------------- kernel 2: f32 -> bf16 convert ----------------
__global__ void cvt_bf16_kernel(const float* __restrict__ in,
                                unsigned short* __restrict__ out, int n) {
  int i = blockIdx.x * 256 + threadIdx.x;
  if (i < n) out[i] = f2bf(in[i]);
}

// ---------------- kernel 3: V [B,NK,VD] f32 -> vT [B,VD,NK] bf16 ----------------
__global__ void vtranspose_bf16_kernel(const float* __restrict__ v,
                                       unsigned short* __restrict__ vt) {
  __shared__ float t[32][33];
  int b = blockIdx.z, k0 = blockIdx.x * 32, n0 = blockIdx.y * 32;
  int tx = threadIdx.x & 31, ty = threadIdx.x >> 5;   // ty in 0..7
  const float* src = v + (size_t)b * NK * VD;
  #pragma unroll
  for (int i = 0; i < 4; i++) {
    int r = ty + 8 * i;
    t[r][tx] = src[(size_t)(k0 + r) * VD + n0 + tx];
  }
  __syncthreads();
  unsigned short* dst = vt + (size_t)b * VD * NK;
  #pragma unroll
  for (int i = 0; i < 4; i++) {
    int a = ty + 8 * i;
    dst[(size_t)(n0 + a) * NK + k0 + tx] = f2bf(t[tx][a]);
  }
}

// ---------------- kernel 4: bf16 GEMM with bias: Out[M,N] = A[M,K] @ W[N,K]^T + bias ----------------
__global__ void gemm_bias_bf16_kernel(const unsigned short* __restrict__ A,
                                      const unsigned short* __restrict__ W,
                                      const float* __restrict__ bias,
                                      unsigned short* __restrict__ out,
                                      int M, int N, int K) {
  int lane = threadIdx.x & 31, wave = threadIdx.x >> 5;
  int m0 = blockIdx.x * 16;
  int n0 = blockIdx.y * 128 + wave * 16;
  const unsigned short* Arow = A + (size_t)(m0 + (lane & 15)) * K;
  const unsigned short* Wrow = W + (size_t)(n0 + (lane & 15)) * K;
  int loA = (lane < 16) ? 0 : 16;
  int loB = (lane < 16) ? 0 : 16;
  v8f acc = {};
  for (int k = 0; k < K; k += 32) {
    __builtin_prefetch((const void*)(Arow + k + 256), 0, 1);
    __builtin_prefetch((const void*)(Wrow + k + 256), 0, 1);
    v16bf a = load_a_bf16(Arow, k, loA);
    v16bf b = *(const v16bf*)(Wrow + k + loB);
    acc = __builtin_amdgcn_wmma_f32_16x16x32_bf16(false, a, false, b,
                                                  (short)0, acc, false, false);
  }
  float bv = bias[n0 + (lane & 15)];
  int rbase = (lane < 16) ? 0 : 8;      // C layout: lanes>=16 hold rows 8..15
  #pragma unroll
  for (int r = 0; r < 8; r++)
    out[(size_t)(m0 + rbase + r) * N + n0 + (lane & 15)] = f2bf(acc[r] + bv);
}

// ---------------- kernel 5: fused attention ----------------
// One block per (b, 16-row q-tile). 8 waves; each wave owns heads {wave, wave+8}.
// Q tile arrives via one TDM DMA; K tiles stream through a per-wave double-buffered
// TDM ring in LDS (s_wait_tensorcnt pipelined). Two-pass flash softmax; normalized
// probs accumulated with ds_add_f32; attn_mean written; out = attn_mean @ V (WMMA).
__global__ void attention_kernel(const unsigned short* __restrict__ Qp,
                                 const unsigned short* __restrict__ Kp,
                                 const unsigned short* __restrict__ vT,
                                 float* __restrict__ out,        // [B,NQ,VD]
                                 float* __restrict__ attn_out) { // [B,NQ,NK]
  extern __shared__ char smem[];
  unsigned short* qtile  = (unsigned short*)smem;                              // [16][D] bf16, 32 KB
  float* attn            = (float*)(smem + (size_t)16 * D * 2);                // [16][NK] f32, 128 KB
  unsigned short* kstage = (unsigned short*)(smem + (size_t)16 * D * 2
                                                  + (size_t)16 * NK * 4);      // 8 waves x 2 x 16*HD, 32 KB
  constexpr int KTILE = 16 * HD;            // elems per K tile (2 KB)

  int bq = blockIdx.x;
  int b  = bq >> 7;               // NQ/16 == 128 tiles per batch
  int q0 = (bq & 127) * 16;
  int tid = threadIdx.x, lane = tid & 31, wave = tid >> 5;

  const unsigned short* Qbase = Qp + ((size_t)b * NQ + q0) * D;

#if HAVE_TDM
  if (wave == 0) {
    unsigned qtile_lds = (unsigned)(uintptr_t)(void*)qtile;
    tdm_load_2d_bf16(qtile_lds, Qbase, /*w*/D, /*h*/16, /*stride*/D);
    __builtin_amdgcn_s_wait_tensorcnt(0);
  }
#else
  for (int i = tid; i < 16 * D / 8; i += 256)
    ((uint4*)qtile)[i] = ((const uint4*)Qbase)[i];
#endif
  for (int i = tid; i < 16 * NK; i += 256) attn[i] = 0.f;
  __syncthreads();

  int loA = (lane < 16) ? 0 : 16;
  int loB = (lane < 16) ? 0 : 16;
  int rbase = (lane < 16) ? 0 : 8;
  const float scale = 0.125f;     // HD^-0.5
  const unsigned short* Kb = Kp + (size_t)b * NK * D;
  unsigned short* kbuf = kstage + (size_t)wave * 2 * KTILE;
#if HAVE_TDM
  unsigned kbuf_lds = (unsigned)(uintptr_t)(void*)kbuf;
#endif

  for (int hh = 0; hh < 2; ++hh) {
    int h = wave + hh * 8;
    int hoff = h * HD;
    const unsigned short* qrow = qtile + (size_t)(lane & 15) * D + hoff;

    float m[8], l[8];
    #pragma unroll
    for (int r = 0; r < 8; r++) { m[r] = -1e30f; l[r] = 0.f; }

    for (int pass = 0; pass < 2; ++pass) {
      float inv_l[8];
      if (pass == 1) {
        #pragma unroll
        for (int r = 0; r < 8; r++) inv_l[r] = 1.f / l[r];
      }
#if HAVE_TDM
      // prologue: kick tile 0 into buffer 0
      tdm_load_2d_bf16(kbuf_lds, Kb + (size_t)0 * D + hoff, HD, 16, D);
#endif
      for (int kt = 0; kt < NK / 16; ++kt) {
        int k0 = kt * 16;
        v8f acc = {};
#if HAVE_TDM
        if (kt + 1 < NK / 16) {   // issue next tile, wait for current
          tdm_load_2d_bf16(kbuf_lds + ((kt + 1) & 1) * (KTILE * 2),
                           Kb + (size_t)(k0 + 16) * D + hoff, HD, 16, D);
          __builtin_amdgcn_s_wait_tensorcnt(1);
        } else {
          __builtin_amdgcn_s_wait_tensorcnt(0);
        }
        const unsigned short* krow = kbuf + (kt & 1) * KTILE + (size_t)(lane & 15) * HD;
        #pragma unroll
        for (int kk = 0; kk < HD; kk += 32) {
          v16bf a  = load_a_bf16(qrow, kk, loA);
          v16bf bm = *(const v16bf*)(krow + kk + loB);   // ds_load from staged tile
          acc = __builtin_amdgcn_wmma_f32_16x16x32_bf16(false, a, false, bm,
                                                        (short)0, acc, false, false);
        }
#else
        const unsigned short* krow = Kb + (size_t)(k0 + (lane & 15)) * D + hoff;
        #pragma unroll
        for (int kk = 0; kk < HD; kk += 32) {
          v16bf a  = load_a_bf16(qrow, kk, loA);
          v16bf bm = *(const v16bf*)(krow + kk + loB);
          acc = __builtin_amdgcn_wmma_f32_16x16x32_bf16(false, a, false, bm,
                                                        (short)0, acc, false, false);
        }
#endif
        if (pass == 0) {
          #pragma unroll
          for (int r = 0; r < 8; r++) {
            float s    = acc[r] * scale;
            float tmax = warp16_max(s);
            float mn   = fmaxf(m[r], tmax);
            float p    = __expf(s - mn);
            float psum = warp16_sum(p);
            l[r] = l[r] * __expf(m[r] - mn) + psum;
            m[r] = mn;
          }
        } else {
          int col = k0 + (lane & 15);
          #pragma unroll
          for (int r = 0; r < 8; r++) {
            float p = __expf(acc[r] * scale - m[r]) * inv_l[r];
            atomicAdd(&attn[(size_t)(rbase + r) * NK + col], p);   // ds_add_f32
          }
        }
      }
    }
  }
  __syncthreads();

  // ---- finalize attn_mean (divide by H), write to output ----
  float* attn_row_out = attn_out + ((size_t)b * NQ + q0) * NK;
  for (int i = tid; i < 16 * NK; i += 256) {
    float v = attn[i] * (1.f / H);
    attn[i] = v;
    attn_row_out[i] = v;
  }
  __syncthreads();

  // ---- out(16 x VD) = attn_mean(16 x NK) @ V(NK x VD); wave owns 64 columns ----
  const unsigned short* vTb = vT + (size_t)b * VD * NK;
  int n0 = wave * 64;
  v8f acc4[4] = {};
  const float* arow = attn + (size_t)(lane & 15) * NK;
  for (int k = 0; k < NK; k += 32) {
    union { v16bf v; unsigned short u[16]; } afr;
    #pragma unroll
    for (int j = 0; j < 8; j++) afr.u[j]     = f2bf(arow[k + loA + j]);
    #pragma unroll
    for (int j = 0; j < 8; j++) afr.u[8 + j] = f2bf(arow[k + loA + 8 + j]);
    #pragma unroll
    for (int nt = 0; nt < 4; nt++) {
      const unsigned short* vrow = vTb + (size_t)(n0 + nt * 16 + (lane & 15)) * NK;
      v16bf bm = *(const v16bf*)(vrow + k + loB);
      acc4[nt] = __builtin_amdgcn_wmma_f32_16x16x32_bf16(false, afr.v, false, bm,
                                                         (short)0, acc4[nt], false, false);
    }
  }
  float* orow_base = out + ((size_t)b * NQ + q0) * VD;
  #pragma unroll
  for (int nt = 0; nt < 4; nt++) {
    int col = n0 + nt * 16 + (lane & 15);
    #pragma unroll
    for (int r = 0; r < 8; r++)
      orow_base[(size_t)(rbase + r) * VD + col] = acc4[nt][r];
  }
}

// ---------------- host launch ----------------
extern "C" void kernel_launch(void* const* d_in, const int* in_sizes, int n_in,
                              void* d_out, int out_size, void* d_ws, size_t ws_size,
                              hipStream_t stream) {
  const float* query    = (const float*)d_in[0];   // [B,NQ,D]
  const float* key      = (const float*)d_in[1];   // [B,NK,D]
  const float* value    = (const float*)d_in[2];   // [B,NK,VD]
  const float* norm_q_w = (const float*)d_in[3];   // [D]
  const float* norm_k_w = (const float*)d_in[4];   // [D]
  const float* q_proj_w = (const float*)d_in[5];   // [D,D]
  const float* k_proj_w = (const float*)d_in[6];   // [D,D]
  const float* q_bias   = (const float*)d_in[7];   // [D]
  const float* k_bias   = (const float*)d_in[8];   // [D]

  float* out_main = (float*)d_out;                             // [B,NQ,VD]
  float* out_attn = (float*)d_out + (size_t)B * NQ * VD;       // [B,NQ,NK]

  // workspace layout (bf16 stored as ushort)
  char* ws = (char*)d_ws;
  size_t off = 0;
  auto carve = [&](size_t bytes) { char* p = ws + off; off += (bytes + 255) & ~(size_t)255; return p; };
  unsigned short* qn = (unsigned short*)carve((size_t)B * NQ * D * 2);
  unsigned short* kn = (unsigned short*)carve((size_t)B * NK * D * 2);
  unsigned short* wq = (unsigned short*)carve((size_t)D * D * 2);
  unsigned short* wk = (unsigned short*)carve((size_t)D * D * 2);
  unsigned short* vt = (unsigned short*)carve((size_t)B * VD * NK * 2);
  unsigned short* qp = (unsigned short*)carve((size_t)B * NQ * D * 2);
  unsigned short* kp = (unsigned short*)carve((size_t)B * NK * D * 2);

  // 1) RMSNorm -> bf16
  rmsnorm_bf16_kernel<<<B * NQ, 256, 0, stream>>>(query, norm_q_w, qn);
  rmsnorm_bf16_kernel<<<B * NK, 256, 0, stream>>>(key,   norm_k_w, kn);

  // 2) weights -> bf16
  cvt_bf16_kernel<<<(D * D + 255) / 256, 256, 0, stream>>>(q_proj_w, wq, D * D);
  cvt_bf16_kernel<<<(D * D + 255) / 256, 256, 0, stream>>>(k_proj_w, wk, D * D);

  // 3) V -> vT bf16 [B,VD,NK]
  vtranspose_bf16_kernel<<<dim3(NK / 32, VD / 32, B), 256, 0, stream>>>(value, vt);

  // 4) projections: qp = qn @ Wq^T + qb ; kp = kn @ Wk^T + kb
  gemm_bias_bf16_kernel<<<dim3((B * NQ) / 16, D / 128), 256, 0, stream>>>(qn, wq, q_bias, qp, B * NQ, D, D);
  gemm_bias_bf16_kernel<<<dim3((B * NK) / 16, D / 128), 256, 0, stream>>>(kn, wk, k_bias, kp, B * NK, D, D);

  // 5) fused attention (192 KB dynamic LDS: 32K qtile + 128K attn + 32K K-stage)
  size_t shmem = (size_t)16 * D * 2 + (size_t)16 * NK * 4 + (size_t)8 * 2 * 16 * HD * 2;
  attention_kernel<<<B * (NQ / 16), 256, shmem, stream>>>(qp, kp, vt, out_main, out_attn);
}
